// CBottleneck_38963943309892
// MI455X (gfx1250) — matrix operand order repaired and checked by reading, not compiled, not run on hardware
//
#include <hip/hip_runtime.h>

// ---------------- problem constants ----------------
#define B_  2
#define E_  512
#define C_  4
#define T_  1001
#define H_  8
#define D_  64
#define HD_ 256
#define CE_ (C_*E_)

typedef __attribute__((ext_vector_type(16))) _Float16 v16h;
typedef __attribute__((ext_vector_type(8)))  _Float16 v8h;
typedef __attribute__((ext_vector_type(8)))  float    v8f;

union F16x16 { v16h v; v8h v8[2]; _Float16 h[16]; };
union F32x8  { v8f  v; float    f[8];  };

// ---------------------------------------------------------------------------
// Batched complex GEMM:  C[m,n] = scale * sum_k A[m,k]*B[k,n] (+bias)(+mask)(CReLU)
// A,B,C complex interleaved (re,im) float. transB=1 means B given as N x K.
// Complex product via 4 real WMMAs per tile:
//   Cr += Ar*Br + Ai*(-Bi) ; Ci += Ar*Bi + Ai*Br
// LDS layouts give every lane's fragment as two aligned 16B runs
// (ds_load_b128 pairs). Interior blocks take an unconditional, batched
// global-load fast path so HBM latency is pipelined, not serialized.
// ---------------------------------------------------------------------------
__global__ __launch_bounds__(256) void cgemm_wmma(
    const float* __restrict__ A, const float* __restrict__ Bm,
    const float* __restrict__ bias, const float* __restrict__ maskM,
    float* __restrict__ Cm,
    int M, int N, int K, int lda, int ldb, int ldc,
    long aS1, long aS2, long bS1, long bS2, long cS1, long cS2,
    int batchDiv, int transB, int hasBias, float scale, int doCrelu, int addMask)
{
  __shared__ __align__(16) _Float16 lAr[64][40];  // [m][k], k-runs aligned
  __shared__ __align__(16) _Float16 lAi[64][40];
  __shared__ __align__(16) _Float16 lBr[64][40];  // [n][k]  (N-major!)
  __shared__ __align__(16) _Float16 lBi[64][40];

  int z  = blockIdx.z;
  int z1 = z / batchDiv, z2 = z % batchDiv;
  A  += 2 * (aS1 * (long)z1 + aS2 * (long)z2);
  Bm += 2 * (bS1 * (long)z1 + bS2 * (long)z2);
  Cm += 2 * (cS1 * (long)z1 + cS2 * (long)z2);
  const float* maskP = maskM ? maskM + 2 * (cS1 * (long)z1 + cS2 * (long)z2) : nullptr;

  int m0 = blockIdx.y * 64;
  int n0 = blockIdx.x * 64;
  int tid  = threadIdx.x;
  int lane = tid & 31, wave = tid >> 5;
  int wm = (wave >> 2) * 32;   // 0 or 32
  int wn = (wave & 3) * 16;    // 0,16,32,48
  int lo = lane & 15, hi = lane >> 4;

  const bool fullM = (m0 + 64 <= M);
  const bool fullN = (n0 + 64 <= N);

  // per-thread staging coordinates
  const int ra = tid >> 5, ka = tid & 31;   // A: row ra+8j? no: row from idx
  (void)ra; (void)ka;

  v8f accR0 = {}, accI0 = {}, accR1 = {}, accI1 = {};

  for (int k0 = 0; k0 < K; k0 += 32) {
    const bool fullK = (k0 + 32 <= K);

    // ---- stage A tile 64x32 ----
    if (fullM && fullK) {
      float2 av[8];
      #pragma unroll
      for (int j = 0; j < 8; ++j) {
        int idx = tid + j * 256;
        int r = idx >> 5, kk = idx & 31;
        av[j] = *(const float2*)&A[2 * ((long)(m0 + r) * lda + (k0 + kk))];
      }
      #pragma unroll
      for (int j = 0; j < 8; ++j) {
        int idx = tid + j * 256;
        int r = idx >> 5, kk = idx & 31;
        lAr[r][kk] = (_Float16)av[j].x;
        lAi[r][kk] = (_Float16)av[j].y;
      }
    } else {
      #pragma unroll
      for (int j = 0; j < 8; ++j) {
        int idx = tid + j * 256;
        int r = idx >> 5, kk = idx & 31;
        int m = m0 + r, k = k0 + kk;
        float re = 0.f, im = 0.f;
        if (m < M && k < K) {
          long off = 2 * ((long)m * lda + k);
          re = A[off]; im = A[off + 1];
        }
        lAr[r][kk] = (_Float16)re;
        lAi[r][kk] = (_Float16)im;
      }
    }

    // ---- stage B tile 32x64 into N-major LDS: lB[n][k] ----
    if (fullN && fullK) {
      float2 bv[8];
      #pragma unroll
      for (int j = 0; j < 8; ++j) {
        int idx = tid + j * 256;
        int kk = idx >> 6, n = idx & 63;
        long off = transB ? 2 * ((long)(n0 + n) * ldb + (k0 + kk))
                          : 2 * ((long)(k0 + kk) * ldb + (n0 + n));
        bv[j] = *(const float2*)&Bm[off];
      }
      #pragma unroll
      for (int j = 0; j < 8; ++j) {
        int idx = tid + j * 256;
        int kk = idx >> 6, n = idx & 63;
        lBr[n][kk] = (_Float16)bv[j].x;
        lBi[n][kk] = (_Float16)bv[j].y;
      }
    } else {
      #pragma unroll
      for (int j = 0; j < 8; ++j) {
        int idx = tid + j * 256;
        int kk = idx >> 6, n = idx & 63;
        int k = k0 + kk, nn = n0 + n;
        float re = 0.f, im = 0.f;
        if (k < K && nn < N) {
          long off = transB ? 2 * ((long)nn * ldb + k) : 2 * ((long)k * ldb + nn);
          re = Bm[off]; im = Bm[off + 1];
        }
        lBr[n][kk] = (_Float16)re;
        lBi[n][kk] = (_Float16)im;
      }
    }

    // prefetch next A K-tile into L2 (global_prefetch_b8)
    if (k0 + 32 < K) {
      int pm = m0 + (tid >> 2);
      if (pm < M) __builtin_prefetch(&A[2 * ((long)pm * lda + k0 + 32)], 0, 1);
    }
    __syncthreads();

    // vector fragment loads: two aligned b128 per fragment
    F16x16 ar0, ai0, ar1, ai1, br, bi, bin;
    ar0.v8[0] = *(const v8h*)&lAr[wm + lo][8 * hi];
    ar0.v8[1] = *(const v8h*)&lAr[wm + lo][16 + 8 * hi];
    ai0.v8[0] = *(const v8h*)&lAi[wm + lo][8 * hi];
    ai0.v8[1] = *(const v8h*)&lAi[wm + lo][16 + 8 * hi];
    ar1.v8[0] = *(const v8h*)&lAr[wm + 16 + lo][8 * hi];
    ar1.v8[1] = *(const v8h*)&lAr[wm + 16 + lo][16 + 8 * hi];
    ai1.v8[0] = *(const v8h*)&lAi[wm + 16 + lo][8 * hi];
    ai1.v8[1] = *(const v8h*)&lAi[wm + 16 + lo][16 + 8 * hi];
    br.v8[0]  = *(const v8h*)&lBr[wn + lo][8 * hi];
    br.v8[1]  = *(const v8h*)&lBr[wn + lo][16 + 8 * hi];
    bi.v8[0]  = *(const v8h*)&lBi[wn + lo][8 * hi];
    bi.v8[1]  = *(const v8h*)&lBi[wn + lo][16 + 8 * hi];
    bin.v = -bi.v;  // -Bi fragment (packed sign flip)

    accR0 = __builtin_amdgcn_wmma_f32_16x16x32_f16(false, ar0.v, false, br.v,  (short)0, accR0, false, false);
    accR0 = __builtin_amdgcn_wmma_f32_16x16x32_f16(false, ai0.v, false, bin.v, (short)0, accR0, false, false);
    accI0 = __builtin_amdgcn_wmma_f32_16x16x32_f16(false, ar0.v, false, bi.v,  (short)0, accI0, false, false);
    accI0 = __builtin_amdgcn_wmma_f32_16x16x32_f16(false, ai0.v, false, br.v,  (short)0, accI0, false, false);
    accR1 = __builtin_amdgcn_wmma_f32_16x16x32_f16(false, ar1.v, false, br.v,  (short)0, accR1, false, false);
    accR1 = __builtin_amdgcn_wmma_f32_16x16x32_f16(false, ai1.v, false, bin.v, (short)0, accR1, false, false);
    accI1 = __builtin_amdgcn_wmma_f32_16x16x32_f16(false, ar1.v, false, bi.v,  (short)0, accI1, false, false);
    accI1 = __builtin_amdgcn_wmma_f32_16x16x32_f16(false, ai1.v, false, br.v,  (short)0, accI1, false, false);
    __syncthreads();
  }

  // epilogue: C/D layout: VGPR r -> M = r + 8*hi ; N = lane&15
  #pragma unroll
  for (int t = 0; t < 2; ++t) {
    F32x8 rr, ii;
    rr.v = t ? accR1 : accR0;
    ii.v = t ? accI1 : accI0;
    #pragma unroll
    for (int r = 0; r < 8; ++r) {
      int m = m0 + wm + 16 * t + r + 8 * hi;
      int n = n0 + wn + lo;
      if (m < M && n < N) {
        long off = 2 * ((long)m * ldc + n);
        float vr = rr.f[r] * scale;
        float vi = ii.f[r] * scale;
        if (hasBias) { vr += bias[2 * n]; vi += bias[2 * n + 1]; }
        if (addMask) { vr += maskP[off]; vi += maskP[off + 1]; }
        if (doCrelu) { vr = vr > 0.f ? vr : 0.f; vi = vi > 0.f ? vi : 0.f; }
        Cm[off] = vr; Cm[off + 1] = vi;
      }
    }
  }
}

// ----- input transpose: emb (B,E,C,T) re/im planes -> (B,T,C*E) interleaved -----
__global__ void pack_x(const float* __restrict__ re, const float* __restrict__ im,
                       float* __restrict__ xin, long total) {
  long i = blockIdx.x * (long)blockDim.x + threadIdx.x;
  if (i >= total) return;
  int e = (int)(i % E_);
  int c = (int)((i / E_) % C_);
  int t = (int)((i / (E_ * C_)) % T_);
  int b = (int)(i / ((long)E_ * C_ * T_));
  long src = ((((long)b * E_ + e) * C_ + c) * T_) + t;
  xin[2 * i]     = re[src];
  xin[2 * i + 1] = im[src];
}

// ----- interleave hrtf re/im planes -----
__global__ void pack_cplx(const float* __restrict__ re, const float* __restrict__ im,
                          float* __restrict__ out, long total) {
  long i = blockIdx.x * (long)blockDim.x + threadIdx.x;
  if (i >= total) return;
  out[2 * i] = re[i]; out[2 * i + 1] = im[i];
}

// ----- tiny HRTF attention core (seq=2, dim=256, h=8, d=32), scalar -----
__global__ void hrtf_attn(const float* __restrict__ q, const float* __restrict__ k,
                          const float* __restrict__ v, float* __restrict__ o) {
  long i = blockIdx.x * (long)blockDim.x + threadIdx.x;  // B*2*HD
  if (i >= (long)B_ * 2 * HD_) return;
  int e = (int)(i % HD_);
  int t = (int)((i / HD_) % 2);
  int b = (int)(i / (2 * HD_));
  int h = e / 32;
  const float scale = 0.17677669529663687f;  // 32^-0.5
  float orr = 0.f, oii = 0.f;
  for (int s = 0; s < 2; ++s) {
    float sr = 0.f, si = 0.f;
    for (int d = 0; d < 32; ++d) {
      long qo = 2 * (((long)b * 2 + t) * HD_ + h * 32 + d);
      long ko = 2 * (((long)b * 2 + s) * HD_ + h * 32 + d);
      float qr = q[qo], qi = q[qo + 1], kr = k[ko], ki = k[ko + 1];
      sr += qr * kr - qi * ki;
      si += qr * ki + qi * kr;
    }
    sr *= scale; si *= scale;
    sr = sr > 0.f ? sr : 0.f;
    si = si > 0.f ? si : 0.f;
    long vo = 2 * (((long)b * 2 + s) * HD_ + e);
    float vr = v[vo], vi = v[vo + 1];
    orr += sr * vr - si * vi;
    oii += sr * vi + si * vr;
  }
  o[2 * i] = orr; o[2 * i + 1] = oii;
}

// ----- global magnitude layernorm over (2,256) per batch, gamma/beta per dim-1 -----
__global__ void globln_hrtf(float* __restrict__ x, const float* __restrict__ g,
                            const float* __restrict__ bt) {
  __shared__ float s1[256], s2[256];
  int b = blockIdx.x, tid = threadIdx.x;
  float sum = 0.f, sq = 0.f;
  for (int i = tid; i < 512; i += 256) {
    long off = 2 * ((long)b * 512 + i);
    float m = sqrtf(x[off] * x[off] + x[off + 1] * x[off + 1]);
    sum += m; sq += m * m;
  }
  s1[tid] = sum; s2[tid] = sq; __syncthreads();
  for (int st = 128; st > 0; st >>= 1) {
    if (tid < st) { s1[tid] += s1[tid + st]; s2[tid] += s2[tid + st]; }
    __syncthreads();
  }
  float mean = s1[0] / 512.f;
  float var  = s2[0] / 512.f - mean * mean;
  float inv  = rsqrtf(var + 1e-8f);
  for (int i = tid; i < 512; i += 256) {
    long off = 2 * ((long)b * 512 + i);
    float xr = x[off], xi = x[off + 1];
    float mag = sqrtf(xr * xr + xi * xi);
    float nm = (mag - mean) * inv;
    float sc = mag > 0.f ? nm / mag : 0.f;
    float nr = xr * sc, ni = xi * sc;
    int ch = i >> 8;  // dim-1 channel (0 or 1)
    float gr = g[2 * ch], gi = g[2 * ch + 1], br = bt[2 * ch], bi = bt[2 * ch + 1];
    x[off]     = nr * gr - ni * gi + br;
    x[off + 1] = nr * gi + ni * gr + bi;
  }
}

// ----- bn = x * h (broadcast over T), complex -----
__global__ void bmul(const float* __restrict__ x, const float* __restrict__ hv,
                     float* __restrict__ y, long total) {
  long i = blockIdx.x * (long)blockDim.x + threadIdx.x;
  if (i >= total) return;
  int e = (int)(i % E_);
  int b = (int)(i / ((long)E_ * T_));
  long xo = 2 * i, ho = 2 * ((long)b * E_ + e);
  float xr = x[xo], xi = x[xo + 1], hr = hv[ho], hi2 = hv[ho + 1];
  y[xo]     = xr * hr - xi * hi2;
  y[xo + 1] = xr * hi2 + xi * hr;
}

// ----- globln over (T,E) per batch: stats pass -----
__global__ void gln_stats(const float* __restrict__ x, float* __restrict__ stats) {
  __shared__ float s1[256], s2[256];
  int b = blockIdx.x, tid = threadIdx.x;
  long n = (long)T_ * E_;
  float sum = 0.f, sq = 0.f;
  for (long i = tid; i < n; i += 256) {
    long off = 2 * ((long)b * n + i);
    float m = sqrtf(x[off] * x[off] + x[off + 1] * x[off + 1]);
    sum += m; sq += m * m;
  }
  s1[tid] = sum; s2[tid] = sq; __syncthreads();
  for (int st = 128; st > 0; st >>= 1) {
    if (tid < st) { s1[tid] += s1[tid + st]; s2[tid] += s2[tid + st]; }
    __syncthreads();
  }
  if (tid == 0) {
    float mean = s1[0] / (float)n;
    float var  = s2[0] / (float)n - mean * mean;
    stats[2 * b] = mean;
    stats[2 * b + 1] = rsqrtf(var + 1e-8f);
  }
}

// ----- globln apply: y = norm(x)*gamma[t] + beta[t] -----
__global__ void gln_apply(const float* __restrict__ x, const float* __restrict__ stats,
                          const float* __restrict__ g, const float* __restrict__ bt,
                          float* __restrict__ y, long total) {
  long i = blockIdx.x * (long)blockDim.x + threadIdx.x;
  if (i >= total) return;
  long btm = i / E_;
  int t = (int)(btm % T_);
  int b = (int)(btm / T_);
  float mean = stats[2 * b], inv = stats[2 * b + 1];
  long off = 2 * i;
  float xr = x[off], xi = x[off + 1];
  float mag = sqrtf(xr * xr + xi * xi);
  float nm = (mag - mean) * inv;
  float sc = mag > 0.f ? nm / mag : 0.f;
  float nr = xr * sc, ni = xi * sc;
  float gr = g[2 * t], gi = g[2 * t + 1], br = bt[2 * t], bi = bt[2 * t + 1];
  y[off]     = nr * gr - ni * gi + br;
  y[off + 1] = nr * gi + ni * gr + bi;
}

// ----- output transpose: (B,T,E*C) -> (B,E,C,T) interleaved complex -----
__global__ void pack_out(const float* __restrict__ y, float* __restrict__ out, long total) {
  long i = blockIdx.x * (long)blockDim.x + threadIdx.x;
  if (i >= total) return;
  int t = (int)(i % T_);
  int c = (int)((i / T_) % C_);
  int e = (int)((i / ((long)T_ * C_)) % E_);
  int b = (int)(i / ((long)T_ * C_ * E_));
  long src = 2 * ((((long)b * T_ + t) * CE_) + e * C_ + c);
  out[2 * i]     = y[src];
  out[2 * i + 1] = y[src + 1];
}

// ---------------------------------------------------------------------------
static inline void launch_cgemm(hipStream_t st, const float* A, const float* Bm,
                                const float* bias, const float* mask, float* Cm,
                                int M, int N, int K, int lda, int ldb, int ldc,
                                long aS1, long aS2, long bS1, long bS2, long cS1, long cS2,
                                int batches, int batchDiv, int transB,
                                float scale, int crelu, int addMask) {
  dim3 grid((N + 63) / 64, (M + 63) / 64, batches);
  cgemm_wmma<<<grid, 256, 0, st>>>(A, Bm, bias, mask, Cm, M, N, K, lda, ldb, ldc,
                                   aS1, aS2, bS1, bS2, cS1, cS2, batchDiv, transB,
                                   bias != nullptr, scale, crelu, addMask);
}

extern "C" void kernel_launch(void* const* d_in, const int* in_sizes, int n_in,
                              void* d_out, int out_size, void* d_ws, size_t ws_size,
                              hipStream_t stream) {
  auto P = [&](int i) { return (const float*)d_in[i]; };
  const float* emb_re = P(0);
  const float* emb_im = P(1);
  const float* hr_re  = P(2);
  const float* hr_im  = P(3);
  // params (complex interleaved), flattened in dict insertion order:
  // 4 fc1_w, 5 fc1_b, 6..13 sa_mix, 14..21 sa0, 22..29 sa1, 30..37 sa2,
  // 38..45 sa3, 46..53 sa_hrtf, 54 fc_hrtf_w, 55 fc_hrtf_b, 56 op_w, 57 op_b,
  // 58 gln_hrtf_g, 59 gln_hrtf_b, 60..67 gln{0..3}_{g,b}

  float* ws = (float*)d_ws;
  const size_t SB   = (size_t)B_ * T_ * E_ * 2;       // (B,T,E) complex
  const size_t SBIG = (size_t)B_ * T_ * CE_ * 2;      // (B,T,2048) complex
  const size_t SS   = (size_t)B_ * H_ * T_ * T_ * 2;  // scores complex
  float* XIN  = ws;            // doubles as final projection output
  float* X1   = XIN + SBIG;
  float* X2   = X1 + SB;
  float* Q    = X2 + SB;
  float* Kb   = Q + SB;
  float* V    = Kb + SB;
  float* O    = V + SB;
  float* S    = O + SB;
  float* HIN  = S + SS;
  float* HQ   = HIN + 2048;
  float* HK   = HQ + 2048;
  float* HV   = HK + 2048;
  float* HO   = HV + 2048;
  float* HA   = HO + 2048;
  float* HVEC = HA + 2048;
  float* STATS = HVEC + 2048;

  const int MT = B_ * T_;       // 2002
  const long zero = 0;
  const float scScale = 0.125f; // 64^-0.5

  // 1) pack input (B,E,C,T)->(B,T,C*E)
  {
    long n = (long)B_ * E_ * C_ * T_;
    pack_x<<<(unsigned)((n + 255) / 256), 256, 0, stream>>>(emb_re, emb_im, XIN, n);
  }
  // 2) fc1
  launch_cgemm(stream, XIN, P(4), P(5), nullptr, X1, MT, E_, CE_, CE_, CE_, E_,
               zero, zero, zero, zero, zero, zero, 1, 1, 1, 1.f, 0, 0);

  // attention block helper
  auto attn = [&](const float* xin, float* xout, int pb, int useMask) {
    launch_cgemm(stream, xin, P(pb + 0), P(pb + 1), nullptr, Q,  MT, E_, E_, E_, E_, E_,
                 zero, zero, zero, zero, zero, zero, 1, 1, 1, 1.f, 0, 0);
    launch_cgemm(stream, xin, P(pb + 2), P(pb + 3), nullptr, Kb, MT, E_, E_, E_, E_, E_,
                 zero, zero, zero, zero, zero, zero, 1, 1, 1, 1.f, 0, 0);
    launch_cgemm(stream, xin, P(pb + 4), P(pb + 5), nullptr, V,  MT, E_, E_, E_, E_, E_,
                 zero, zero, zero, zero, zero, zero, 1, 1, 1, 1.f, 0, 0);
    // scores: fused scale + (mask add) + CReLU, in place over S (S is next mask)
    launch_cgemm(stream, Q, Kb, nullptr, useMask ? S : nullptr, S,
                 T_, T_, D_, E_, E_, T_,
                 (long)T_ * E_, (long)D_, (long)T_ * E_, (long)D_,
                 (long)H_ * T_ * T_, (long)T_ * T_,
                 B_ * H_, H_, 1, scScale, 1, useMask);
    // context: O = W * V
    launch_cgemm(stream, S, V, nullptr, nullptr, O,
                 T_, D_, T_, T_, E_, E_,
                 (long)H_ * T_ * T_, (long)T_ * T_, (long)T_ * E_, (long)D_,
                 (long)T_ * E_, (long)D_,
                 B_ * H_, H_, 0, 1.f, 0, 0);
    // output projection
    launch_cgemm(stream, O, P(pb + 6), P(pb + 7), nullptr, xout, MT, E_, E_, E_, E_, E_,
                 zero, zero, zero, zero, zero, zero, 1, 1, 1, 1.f, 0, 0);
  };

  // 3) sa_mix: X1 -> X2 (its attention weights in S are discarded by sa0)
  attn(X1, X2, 6, 0);

  // 4) HRTF path
  {
    long n = (long)B_ * 2 * HD_;
    pack_cplx<<<(unsigned)((n + 255) / 256), 256, 0, stream>>>(hr_re, hr_im, HIN, n);
    launch_cgemm(stream, HIN, P(46), P(47), nullptr, HQ, B_ * 2, HD_, HD_, HD_, HD_, HD_,
                 zero, zero, zero, zero, zero, zero, 1, 1, 1, 1.f, 0, 0);
    launch_cgemm(stream, HIN, P(48), P(49), nullptr, HK, B_ * 2, HD_, HD_, HD_, HD_, HD_,
                 zero, zero, zero, zero, zero, zero, 1, 1, 1, 1.f, 0, 0);
    launch_cgemm(stream, HIN, P(50), P(51), nullptr, HV, B_ * 2, HD_, HD_, HD_, HD_, HD_,
                 zero, zero, zero, zero, zero, zero, 1, 1, 1, 1.f, 0, 0);
    hrtf_attn<<<(unsigned)((n + 255) / 256), 256, 0, stream>>>(HQ, HK, HV, HO);
    launch_cgemm(stream, HO, P(52), P(53), nullptr, HA, B_ * 2, HD_, HD_, HD_, HD_, HD_,
                 zero, zero, zero, zero, zero, zero, 1, 1, 1, 1.f, 0, 0);
    globln_hrtf<<<B_, 256, 0, stream>>>(HA, P(58), P(59));
    // fc_hrtf: (B,512) x (512,512)^T
    launch_cgemm(stream, HA, P(54), P(55), nullptr, HVEC, B_, E_, 2 * HD_,
                 2 * HD_, 2 * HD_, E_,
                 zero, zero, zero, zero, zero, zero, 1, 1, 1, 1.f, 0, 0);
  }

  // 5) bn = x * h (broadcast): X2 * HVEC -> X1
  {
    long n = (long)B_ * T_ * E_;
    bmul<<<(unsigned)((n + 255) / 256), 256, 0, stream>>>(X2, HVEC, X1, n);
  }

  // 6) 4 masked attention blocks + global layernorm
  const int saBase[4] = {14, 22, 30, 38};
  const int glG[4] = {60, 62, 64, 66};
  const int glB[4] = {61, 63, 65, 67};
  for (int i = 0; i < 4; ++i) {
    attn(X1, X2, saBase[i], i > 0 ? 1 : 0);
    gln_stats<<<B_, 256, 0, stream>>>(X2, STATS);
    long n = (long)B_ * T_ * E_;
    gln_apply<<<(unsigned)((n + 255) / 256), 256, 0, stream>>>(X2, STATS, P(glG[i]), P(glB[i]), X1, n);
  }

  // 7) output projection op: (B*T,512) x (2048,512)^T -> XIN (B*T,2048)
  launch_cgemm(stream, X1, P(56), P(57), nullptr, XIN, MT, CE_, E_, E_, E_, CE_,
               zero, zero, zero, zero, zero, zero, 1, 1, 1, 1.f, 0, 0);

  // 8) final transpose to (B,E,C,T) complex interleaved
  {
    long n = (long)B_ * E_ * C_ * T_;
    pack_out<<<(unsigned)((n + 255) / 256), 256, 0, stream>>>(XIN, (float*)d_out, n);
  }
}
